// DoubleNet_59184649338974
// MI455X (gfx1250) — matrix operand order repaired and checked by reading, not compiled, not run on hardware
//
#include <hip/hip_runtime.h>
#include <hip/hip_bf16.h>
#include <math.h>

// ---------------------------------------------------------------------------
// DoubleNet fused inference for MI455X (gfx1250).
// bf16 hi/lo split WMMA GEMMs (layer1 K=256 -> 512 cols, layer2 K=512 -> 17),
// async global->LDS staging, deterministic segment reduction + softmax/NLL.
// ---------------------------------------------------------------------------

typedef __attribute__((ext_vector_type(16))) __bf16 v16bf;
typedef __attribute__((ext_vector_type(8)))  __bf16 v8bf;
typedef __attribute__((ext_vector_type(8)))  float  v8f;

#define HSTR 520   // hidden row stride in ushorts (1040 B: 260 dwords, 4 mod 64 banks)
#define RSTR 260   // raw tile row stride in floats (1040 B)

__device__ __forceinline__ unsigned short f_to_bf16(float f) {
    __bf16 h = (__bf16)f;                       // RNE convert
    return __builtin_bit_cast(unsigned short, h);
}

// ---------------------------------------------------------------------------
// Prep 1: fold BatchNorm into the weight-branch first linear's bias.
// ---------------------------------------------------------------------------
__global__ void prep_bias_kernel(const float* __restrict__ qb1,
                                 const float* __restrict__ wb1,
                                 const float* __restrict__ ww1,
                                 const float* __restrict__ bn_gamma,
                                 const float* __restrict__ bn_beta,
                                 const float* __restrict__ bn_mean,
                                 const float* __restrict__ bn_var,
                                 float* __restrict__ bias) {
    int n = threadIdx.x;                        // 512 threads
    if (n < 256) {
        bias[n] = qb1[n];
    } else {
        int h = n - 256;
        float acc = wb1[h];
        for (int d = 0; d < 256; ++d) {
            float sc = bn_gamma[d] / sqrtf(bn_var[d] + 1e-5f);
            float sh = bn_beta[d] - bn_mean[d] * sc;
            acc += sh * ww1[h * 256 + d];
        }
        bias[n] = acc;
    }
}

// ---------------------------------------------------------------------------
// Prep 2: pack combined layer-1 weights Wc[256 K][512 N] into bf16 hi/lo
// B-fragment order: n = n_tile*16 + (lane&15), k = k_step*32 + (i<8?i:i+8) + (lane<16?0:8)
// ---------------------------------------------------------------------------
__global__ void prep_frags_kernel(const float* __restrict__ qw1,
                                  const float* __restrict__ ww1,
                                  const float* __restrict__ bn_gamma,
                                  const float* __restrict__ bn_var,
                                  unsigned short* __restrict__ w_hi,
                                  unsigned short* __restrict__ w_lo) {
    int fi = blockIdx.x * 256 + threadIdx.x;    // 512 blocks * 256 = 131072
    int i      = fi & 15;
    int lane   = (fi >> 4) & 31;
    int k_step = (fi >> 9) & 7;
    int n_tile = fi >> 12;
    int n = n_tile * 16 + (lane & 15);
    int k = k_step * 32 + ((i < 8) ? i : i + 8) + ((lane < 16) ? 0 : 8);
    float w;
    if (n < 256) {
        w = qw1[n * 256 + k];
    } else {
        float sc = bn_gamma[k] / sqrtf(bn_var[k] + 1e-5f);
        w = ww1[(n - 256) * 256 + k] * sc;
    }
    __bf16 hi = (__bf16)w;
    __bf16 lo = (__bf16)(w - (float)hi);
    w_hi[fi] = __builtin_bit_cast(unsigned short, hi);
    w_lo[fi] = __builtin_bit_cast(unsigned short, lo);
}

// ---------------------------------------------------------------------------
// Prep 3: pack layer-2 weights into B2[512 K(hidden)][32 N] fragments.
//   cols 0..15 : qw2[j][k]   (k < 256, else 0)
//   col  16    : ww2[k-256]  (k >= 256, else 0)
//   cols 17..31: 0
// ---------------------------------------------------------------------------
__global__ void prep_frags2_kernel(const float* __restrict__ qw2,
                                   const float* __restrict__ ww2,
                                   unsigned short* __restrict__ w2_hi,
                                   unsigned short* __restrict__ w2_lo) {
    int fi = blockIdx.x * 256 + threadIdx.x;    // 64 blocks * 256 = 16384
    int i      = fi & 15;
    int lane   = (fi >> 4) & 31;
    int k_step = (fi >> 9) & 15;                // 16 k-steps (K = 512)
    int n_tile = fi >> 13;                      // 2 n-tiles
    int c = n_tile * 16 + (lane & 15);
    int k = k_step * 32 + ((i < 8) ? i : i + 8) + ((lane < 16) ? 0 : 8);
    float w = 0.0f;
    if (c < 16) {
        if (k < 256) w = qw2[c * 256 + k];
    } else if (c == 16) {
        if (k >= 256) w = ww2[k - 256];
    }
    __bf16 hi = (__bf16)w;
    __bf16 lo = (__bf16)(w - (float)hi);
    w2_hi[fi] = __builtin_bit_cast(unsigned short, hi);
    w2_lo[fi] = __builtin_bit_cast(unsigned short, lo);
}

// ---------------------------------------------------------------------------
// Main: one block (8 waves) per segment (50 rows padded to 64).
// ---------------------------------------------------------------------------
__global__ __launch_bounds__(256) void doublenet_main_kernel(
        const float* __restrict__ data,
        const float* __restrict__ qb2,
        const float* __restrict__ wb2,
        const unsigned short* __restrict__ w_hi,
        const unsigned short* __restrict__ w_lo,
        const unsigned short* __restrict__ w2_hi,
        const unsigned short* __restrict__ w2_lo,
        const float* __restrict__ bias,
        const int* __restrict__ tag,
        const int* __restrict__ separates,
        float* __restrict__ out_probs,
        float* __restrict__ lossbuf) {
    // raw fp32 tile (64 x 256, padded rows) unioned with bf16 hidden (64 x 512,
    // padded rows): the raw tile is dead once A-fragments are built.
    __shared__ __align__(32) unsigned char pool[64 * RSTR * 4];      // 66560 B
    // A fragments, split into 16B-stride halves to cut LDS bank conflicts.
    __shared__ __align__(16) unsigned short a_hi0[1024 * 8];         // 16 KB
    __shared__ __align__(16) unsigned short a_hi1[1024 * 8];         // 16 KB
    __shared__ __align__(16) unsigned short a_lo0[1024 * 8];         // 16 KB
    __shared__ __align__(16) unsigned short a_lo1[1024 * 8];         // 16 KB
    __shared__ float bias_s[512];
    __shared__ float qb2_s[16];
    __shared__ float wbuf[64];
    __shared__ float qout[64 * 16];
    __shared__ float segsum[16];
    __shared__ float wb2_s;

    float* raw = reinterpret_cast<float*>(pool);
    unsigned short* hidden = reinterpret_cast<unsigned short*>(pool);

    const int tid = threadIdx.x;
    const int t = blockIdx.x;
    const int seg_start = separates[t];
    const int seg_end = separates[t + 1];
    const int len = seg_end - seg_start;

    for (int idx = tid; idx < 512; idx += 256) bias_s[idx] = bias[idx];
    if (tid < 16) { qb2_s[tid] = qb2[tid]; segsum[tid] = 0.0f; }
    if (tid == 0) wb2_s = wb2[0];

    const int wave = tid >> 5;
    const int lane = tid & 31;

    for (int chunk = 0; chunk < len; chunk += 64) {
        const int rows = (len - chunk < 64) ? (len - chunk) : 64;

        // ---- stage raw tile via async global->LDS (ASYNCcnt path) ----
        for (int it = tid; it < 64 * 64; it += 256) {
            int r  = it >> 6;
            int c4 = (it & 63) << 2;             // float column, 16B chunks
            if (r < rows) {
                const float* g = data + (size_t)(seg_start + chunk + r) * 256 + c4;
                unsigned dst = (unsigned)(size_t)(raw + r * RSTR + c4);
                asm volatile("global_load_async_to_lds_b128 %0, %1, off"
                             :: "v"(dst), "v"(g) : "memory");
            }
        }
        asm volatile("s_wait_asynccnt 0x0" ::: "memory");
        __syncthreads();

        // ---- build A fragments (bf16 hi/lo split) from the raw tile ----
        // 16-bit A 16x32 layout: element i -> k = (i<8 ? i : i+8) + (lane<16?0:8)
        for (int u = tid; u < 1024; u += 256) {
            int l  = u & 31;
            int ks = (u >> 5) & 7;
            int mt = u >> 8;
            int rl = mt * 16 + (l & 15);
            int kbase = ks * 32 + ((l < 16) ? 0 : 8);
            float v[16];
            if (rl < rows) {
                const float* src = raw + rl * RSTR + kbase;
                const float4 f0 = *reinterpret_cast<const float4*>(src + 0);
                const float4 f1 = *reinterpret_cast<const float4*>(src + 4);
                const float4 f2 = *reinterpret_cast<const float4*>(src + 16);
                const float4 f3 = *reinterpret_cast<const float4*>(src + 20);
                v[0]=f0.x; v[1]=f0.y; v[2]=f0.z;  v[3]=f0.w;
                v[4]=f1.x; v[5]=f1.y; v[6]=f1.z;  v[7]=f1.w;
                v[8]=f2.x; v[9]=f2.y; v[10]=f2.z; v[11]=f2.w;
                v[12]=f3.x;v[13]=f3.y;v[14]=f3.z; v[15]=f3.w;
            } else {
                #pragma unroll
                for (int i = 0; i < 16; ++i) v[i] = 0.0f;
            }
            #pragma unroll
            for (int i = 0; i < 16; ++i) {
                __bf16 hi = (__bf16)v[i];
                __bf16 lo = (__bf16)(v[i] - (float)hi);
                unsigned short uhi = __builtin_bit_cast(unsigned short, hi);
                unsigned short ulo = __builtin_bit_cast(unsigned short, lo);
                if (i < 8) { a_hi0[u * 8 + i] = uhi;     a_lo0[u * 8 + i] = ulo; }
                else       { a_hi1[u * 8 + i - 8] = uhi; a_lo1[u * 8 + i - 8] = ulo; }
            }
        }
        __syncthreads();

        // ---- layer-1 GEMM: hidden = relu(data @ Wc + bias), 3-product split ----
        for (int pass = 0; pass < 2; ++pass) {
            const int n0 = wave * 4 + pass * 2;
            v8f acc[4][2];
            #pragma unroll
            for (int m = 0; m < 4; ++m) {
                acc[m][0] = v8f{0.f,0.f,0.f,0.f,0.f,0.f,0.f,0.f};
                acc[m][1] = v8f{0.f,0.f,0.f,0.f,0.f,0.f,0.f,0.f};
            }
            for (int ks = 0; ks < 8; ++ks) {
                union { v16bf v; v8bf h[2]; } ah[4], al[4];
                #pragma unroll
                for (int m = 0; m < 4; ++m) {
                    int u8 = ((m * 8 + ks) * 32 + lane) * 8;
                    ah[m].h[0] = *reinterpret_cast<const v8bf*>(&a_hi0[u8]);
                    ah[m].h[1] = *reinterpret_cast<const v8bf*>(&a_hi1[u8]);
                    al[m].h[0] = *reinterpret_cast<const v8bf*>(&a_lo0[u8]);
                    al[m].h[1] = *reinterpret_cast<const v8bf*>(&a_lo1[u8]);
                }
                #pragma unroll
                for (int nn = 0; nn < 2; ++nn) {
                    int woff = (((n0 + nn) * 8 + ks) * 32 + lane) * 16;
                    v16bf bh = *reinterpret_cast<const v16bf*>(&w_hi[woff]);
                    v16bf bl = *reinterpret_cast<const v16bf*>(&w_lo[woff]);
                    #pragma unroll
                    for (int m = 0; m < 4; ++m) {
                        acc[m][nn] = __builtin_amdgcn_wmma_f32_16x16x32_bf16(
                            false, ah[m].v, false, bh, (short)0, acc[m][nn], false, false);
                        acc[m][nn] = __builtin_amdgcn_wmma_f32_16x16x32_bf16(
                            false, ah[m].v, false, bl, (short)0, acc[m][nn], false, false);
                        acc[m][nn] = __builtin_amdgcn_wmma_f32_16x16x32_bf16(
                            false, al[m].v, false, bh, (short)0, acc[m][nn], false, false);
                    }
                }
            }
            // epilogue: bias + relu, store bf16 hidden (aliases the dead raw tile).
            // C layout: VGPR i -> M = i (lanes 0..15) / i+8 (lanes 16..31), N = lane&15
            #pragma unroll
            for (int nn = 0; nn < 2; ++nn) {
                int ncol = (n0 + nn) * 16 + (lane & 15);
                float b = bias_s[ncol];
                #pragma unroll
                for (int m = 0; m < 4; ++m) {
                    #pragma unroll
                    for (int i = 0; i < 8; ++i) {
                        int row = m * 16 + ((lane < 16) ? i : i + 8);
                        float v = acc[m][nn][i] + b;
                        v = v > 0.0f ? v : 0.0f;
                        hidden[row * HSTR + ncol] = f_to_bf16(v);
                    }
                }
            }
        }
        __syncthreads();

        // ---- layer-2 GEMM via WMMA over K=512 (hidden) ----
        // waves 0..3: quality cols (k-steps 0..7), waves 4..7: weight col (8..15)
        {
            const int nt = wave >> 2;
            const int mt = wave & 3;
            const int ks0 = nt ? 8 : 0;
            v8f acc2 = v8f{0.f,0.f,0.f,0.f,0.f,0.f,0.f,0.f};
            const int arow = mt * 16 + (lane & 15);
            const unsigned short* hbase = &hidden[arow * HSTR + ((lane < 16) ? 0 : 8)];
            #pragma unroll
            for (int kk = 0; kk < 8; ++kk) {
                int ks = ks0 + kk;
                const unsigned short* hp = hbase + ks * 32;
                union { v16bf v; v8bf h[2]; } A;
                A.h[0] = *reinterpret_cast<const v8bf*>(hp);
                A.h[1] = *reinterpret_cast<const v8bf*>(hp + 16);
                int woff = ((nt * 16 + ks) * 32 + lane) * 16;
                v16bf bh = *reinterpret_cast<const v16bf*>(&w2_hi[woff]);
                v16bf bl = *reinterpret_cast<const v16bf*>(&w2_lo[woff]);
                acc2 = __builtin_amdgcn_wmma_f32_16x16x32_bf16(
                    false, A.v, false, bh, (short)0, acc2, false, false);
                acc2 = __builtin_amdgcn_wmma_f32_16x16x32_bf16(
                    false, A.v, false, bl, (short)0, acc2, false, false);
            }
            // weight branch epilogue: only column 0 of n-tile 1 is meaningful
            if (nt == 1 && (lane & 15) == 0) {
                #pragma unroll
                for (int i = 0; i < 8; ++i) {
                    int row = mt * 16 + ((lane < 16) ? i : i + 8);
                    wbuf[row] = 1.0f / (1.0f + expf(-(acc2[i] + wb2_s)));
                }
            }
            __syncthreads();
            // quality epilogue: scale by sigmoid weight
            if (nt == 0) {
                int j = lane & 15;
                float qb = qb2_s[j];
                #pragma unroll
                for (int i = 0; i < 8; ++i) {
                    int row = mt * 16 + ((lane < 16) ? i : i + 8);
                    qout[row * 16 + j] = (acc2[i] + qb) * wbuf[row];
                }
            }
            __syncthreads();
        }

        // ---- deterministic per-segment accumulation ----
        if (tid < 16) {
            float s = segsum[tid];
            for (int r = 0; r < rows; ++r) s += qout[r * 16 + tid];
            segsum[tid] = s;
        }
        __syncthreads();
    }

    // ---- softmax over 16 bins + NLL at tag (tiny; thread 0 serial) ----
    if (tid == 0) {
        float inv_len = 1.0f / (float)len;
        float x[16], mx = -1e30f;
        #pragma unroll
        for (int j = 0; j < 16; ++j) {
            x[j] = segsum[j] * inv_len;
            mx = fmaxf(mx, x[j]);
        }
        float se = 0.0f;
        float e[16];
        #pragma unroll
        for (int j = 0; j < 16; ++j) { e[j] = expf(x[j] - mx); se += e[j]; }
        float inv_se = 1.0f / se;
        #pragma unroll
        for (int j = 0; j < 16; ++j) out_probs[t * 16 + j] = e[j] * inv_se;
        int tg = tag[t];
        lossbuf[t] = (mx + logf(se)) - x[tg & 15];
    }
}

// ---------------------------------------------------------------------------
// Deterministic single-block reduction of per-segment losses.
// ---------------------------------------------------------------------------
__global__ void loss_reduce_kernel(const float* __restrict__ lossbuf, int T,
                                   float* __restrict__ out_loss) {
    __shared__ float red[256];
    float s = 0.0f;
    for (int i = threadIdx.x; i < T; i += 256) s += lossbuf[i];
    red[threadIdx.x] = s;
    __syncthreads();
    for (int off = 128; off > 0; off >>= 1) {
        if (threadIdx.x < off) red[threadIdx.x] += red[threadIdx.x + off];
        __syncthreads();
    }
    if (threadIdx.x == 0) out_loss[0] = red[0] / (float)T;
}

// ---------------------------------------------------------------------------
extern "C" void kernel_launch(void* const* d_in, const int* in_sizes, int n_in,
                              void* d_out, int out_size, void* d_ws, size_t ws_size,
                              hipStream_t stream) {
    (void)n_in; (void)out_size; (void)ws_size;
    const float* data     = (const float*)d_in[0];
    const float* qw1      = (const float*)d_in[1];
    const float* qb1      = (const float*)d_in[2];
    const float* qw2      = (const float*)d_in[3];
    const float* qb2      = (const float*)d_in[4];
    const float* bn_gamma = (const float*)d_in[5];
    const float* bn_beta  = (const float*)d_in[6];
    const float* bn_mean  = (const float*)d_in[7];
    const float* bn_var   = (const float*)d_in[8];
    const float* ww1      = (const float*)d_in[9];
    const float* wb1      = (const float*)d_in[10];
    const float* ww2      = (const float*)d_in[11];
    const float* wb2      = (const float*)d_in[12];
    const int*   tag      = (const int*)d_in[13];
    const int*   seps     = (const int*)d_in[14];

    const int T = in_sizes[13];                  // number of segments

    float* probs = (float*)d_out;                // [T,16]
    float* loss  = probs + (size_t)T * 16;       // scalar

    // workspace layout
    char* ws = (char*)d_ws;
    unsigned short* w_hi  = (unsigned short*)(ws + 0);        // 256 KB
    unsigned short* w_lo  = (unsigned short*)(ws + 262144);   // 256 KB
    float*          bias  = (float*)(ws + 524288);            // 2 KB
    unsigned short* w2_hi = (unsigned short*)(ws + 526336);   // 32 KB
    unsigned short* w2_lo = (unsigned short*)(ws + 559104);   // 32 KB
    float*        lossbuf = (float*)(ws + 591872);            // T*4 B

    prep_bias_kernel<<<1, 512, 0, stream>>>(qb1, wb1, ww1, bn_gamma, bn_beta,
                                            bn_mean, bn_var, bias);
    prep_frags_kernel<<<512, 256, 0, stream>>>(qw1, ww1, bn_gamma, bn_var,
                                               w_hi, w_lo);
    prep_frags2_kernel<<<64, 256, 0, stream>>>(qw2, ww2, w2_hi, w2_lo);
    doublenet_main_kernel<<<T, 256, 0, stream>>>(data, qb2, wb2,
                                                 w_hi, w_lo, w2_hi, w2_lo, bias,
                                                 tag, seps, probs, lossbuf);
    loss_reduce_kernel<<<1, 256, 0, stream>>>(lossbuf, T, loss);
}
